// MambaBlock_8821862826382
// MI455X (gfx1250) — compile-verified, matrix operand order checked by reading, not run on hardware
//
#include <hip/hip_runtime.h>
#include <hip/hip_bf16.h>
#include <math.h>

// ---------------------------------------------------------------------------
// Mamba block forward for MI455X (gfx1250, wave32, WMMA).
//   B=2, T=2048, EMBED=1024, DSTATE=16, DCONV=4, DTRANK=64
// GEMMs: v_wmma_f32_16x16x32_bf16, 128x64 block tile, double-buffered LDS
// filled by GLOBAL_LOAD_ASYNC_TO_LDS_B128 (ASYNCcnt software pipeline).
// Scan: one SSM state per lane, 16-lane shfl_xor reduction per time step.
// ---------------------------------------------------------------------------

static constexpr int B_SZ   = 2;
static constexpr int T_SZ   = 2048;
static constexpr int D_SZ   = 1024;
static constexpr int BT_SZ  = B_SZ * T_SZ;      // 4096
static constexpr int NSTATE = 16;
static constexpr int DTRANK = 64;
static constexpr int SSM_LD = 128;              // 96 padded to 128

typedef __attribute__((ext_vector_type(16))) __bf16        v16bf;
typedef __attribute__((ext_vector_type(8)))  float         v8f;
typedef __attribute__((ext_vector_type(4)))  unsigned int  uv4;

// Async DMA: global -> LDS, 16 bytes per lane, tracked by ASYNCcnt.
__device__ __forceinline__ void async_ld16(unsigned ldsOff, const void* gaddr) {
    asm volatile("global_load_async_to_lds_b128 %0, %1, off"
                 :: "v"(ldsOff), "v"((unsigned long long)gaddr) : "memory");
}
__device__ __forceinline__ void wait_async0() {
    asm volatile("s_wait_asynccnt 0x0" ::: "memory");
}

// ---------------------------------------------------------------------------
// Conversion kernels (fp32 -> bf16)
// ---------------------------------------------------------------------------
__global__ void cvt_f32_bf16(__bf16* __restrict__ dst, const float* __restrict__ src, int n) {
    int i = blockIdx.x * blockDim.x + threadIdx.x;
    if (i < n) dst[i] = (__bf16)src[i];
}

// dst has rowsDst rows; rows >= rowsSrc are zero-filled (tile-aligned padding)
__global__ void cvt_pad_rows(__bf16* __restrict__ dst, const float* __restrict__ src,
                             int rowsDst, int rowsSrc, int cols) {
    int i = blockIdx.x * blockDim.x + threadIdx.x;
    int total = rowsDst * cols;
    if (i >= total) return;
    int row = i / cols, col = i - row * cols;
    dst[i] = (row < rowsSrc) ? (__bf16)src[row * cols + col] : (__bf16)0.0f;
}

// dst (rows, colsDst) <- src (rows, ldSrc) taking first colsDst columns
__global__ void cvt_strided(__bf16* __restrict__ dst, const float* __restrict__ src,
                            int rows, int colsDst, int ldSrc) {
    int i = blockIdx.x * blockDim.x + threadIdx.x;
    int total = rows * colsDst;
    if (i >= total) return;
    int row = i / colsDst, col = i - row * colsDst;
    dst[i] = (__bf16)src[(size_t)row * ldSrc + col];
}

// ---------------------------------------------------------------------------
// Tiled WMMA GEMM:  C[M,N] (f32) = A[M,K] (bf16) * W[N,K]^T (bf16)
// block = 128 threads (4 waves); block tile 128x64; K step 32.
// Each wave owns a 32x64 strip: 2 A-frags x 4 B-frags = 8 WMMAs per K step.
// LDS tiles double-buffered; filled by async global->LDS DMA.
// mode 0: C = acc + bias      mode 1: C = softplus(acc + bias)
// mode 2: C += acc + bias     (bias read only for col < nBias)
// ---------------------------------------------------------------------------
__global__ __launch_bounds__(128) void gemm_bf16_wmma(
    float* __restrict__ C, const __bf16* __restrict__ A, const __bf16* __restrict__ W,
    const float* __restrict__ bias, int M, int N, int K,
    int lda, int ldw, int ldc, int nBias, int mode)
{
    __shared__ __align__(16) __bf16 As[2][128][40];   // +8 half row pad: bank spread
    __shared__ __align__(16) __bf16 Bs[2][64][40];

    const int tid  = threadIdx.x;
    const int lane = tid & 31;
    const int wave = tid >> 5;                        // 0..3 -> 32-row strip
    const int mBase = blockIdx.y * 128;
    const int nBase = blockIdx.x * 64;

    v8f acc[2][4];
    #pragma unroll
    for (int s = 0; s < 2; ++s)
        #pragma unroll
        for (int nt = 0; nt < 4; ++nt) acc[s][nt] = v8f{};

    union Frag { uv4 q[2]; v16bf v; };

    // Stage one K-tile pair (A 128x32, W 64x32) into LDS buffer `buf` via
    // async DMA: 768 16-byte chunks, 6 per thread.
    auto stage = [&](int buf, int k0) {
        #pragma unroll
        for (int c = tid; c < 768; c += 128) {
            if (c < 512) {
                int row = c >> 2, co = (c & 3) * 8;
                async_ld16((unsigned)(size_t)&As[buf][row][co],
                           A + (size_t)(mBase + row) * lda + k0 + co);
            } else {
                int c2 = c - 512;
                int row = c2 >> 2, co = (c2 & 3) * 8;
                async_ld16((unsigned)(size_t)&Bs[buf][row][co],
                           W + (size_t)(nBase + row) * ldw + k0 + co);
            }
        }
    };

    const int nk = K / 32;
    stage(0, 0);
    for (int it = 0; it < nk; ++it) {
        const int cur = it & 1;
        wait_async0();            // my async loads (incl. buf[cur]) landed
        __syncthreads();          // everyone's landed; prev compute on cur^1 done
        if (it + 1 < nk) stage(cur ^ 1, (it + 1) * 32);   // overlap with WMMAs

        // Fragment gather per ISA 16-bit 16x32 layout:
        // lanes 0-15: K 0..7 & 16..23 of row (lane&15); lanes 16-31: K 8..15 & 24..31.
        const int fr = lane & 15;
        const int kh = (lane >> 4) * 8;

        Frag af[2];
        #pragma unroll
        for (int s = 0; s < 2; ++s) {
            af[s].q[0] = *(const uv4*)&As[cur][wave * 32 + s * 16 + fr][kh];
            af[s].q[1] = *(const uv4*)&As[cur][wave * 32 + s * 16 + fr][16 + kh];
        }
        #pragma unroll
        for (int nt = 0; nt < 4; ++nt) {
            Frag bfr;
            bfr.q[0] = *(const uv4*)&Bs[cur][nt * 16 + fr][kh];
            bfr.q[1] = *(const uv4*)&Bs[cur][nt * 16 + fr][16 + kh];
            #pragma unroll
            for (int s = 0; s < 2; ++s)
                acc[s][nt] = __builtin_amdgcn_wmma_f32_16x16x32_bf16(
                    false, af[s].v, false, bfr.v, (short)0, acc[s][nt], false, false);
        }
    }

    // Epilogue. C/D layout: VGPR r -> row r (lanes 0-15) / r+8 (lanes 16-31), col = lane&15.
    const int colLocal = lane & 15;
    const int rowAdd   = (lane >> 4) * 8;
    #pragma unroll
    for (int s = 0; s < 2; ++s) {
        #pragma unroll
        for (int nt = 0; nt < 4; ++nt) {
            int col = nBase + nt * 16 + colLocal;
            float bv = (col < nBias) ? bias[col] : 0.0f;
            #pragma unroll
            for (int r = 0; r < 8; ++r) {
                int row = mBase + wave * 32 + s * 16 + rowAdd + r;
                float v = acc[s][nt][r] + bv;
                float* cp = &C[(size_t)row * ldc + col];
                if (mode == 1) {
                    v = (v > 20.0f) ? v : log1pf(__expf(v));   // softplus
                } else if (mode == 2) {
                    v += *cp;
                }
                *cp = v;
            }
        }
    }
}

// ---------------------------------------------------------------------------
// Depthwise causal conv1d (DCONV=4) + SiLU. Input x_in = xr[:, 0:1024] (ld 2048).
// Writes xc (f32) and xcb (bf16).
// ---------------------------------------------------------------------------
__global__ void conv_silu_kernel(float* __restrict__ xc, __bf16* __restrict__ xcb,
                                 const float* __restrict__ xr,
                                 const float* __restrict__ cw, const float* __restrict__ cb) {
    int i = blockIdx.x * blockDim.x + threadIdx.x;     // 0 .. BT*D-1
    if (i >= BT_SZ * D_SZ) return;
    int d  = i & (D_SZ - 1);
    int bt = i >> 10;
    int t  = bt & (T_SZ - 1);
    float acc = cb[d];
    const float* wrow = cw + d * 4;                    // conv_w (D,1,4)
    #pragma unroll
    for (int j = 0; j < 4; ++j) {
        int tt = t - 3 + j;
        if (tt >= 0)
            acc += wrow[j] * xr[(size_t)(bt - 3 + j) * 2048 + d];
    }
    float s = acc / (1.0f + __expf(-acc));             // silu
    xc[i]  = s;
    xcb[i] = (__bf16)s;
}

// ---------------------------------------------------------------------------
// Selective scan. One (b, d, n) state per lane; 16 lanes (n=0..15) share (b,d).
// h_t = exp(delta*A[n]) * h_{t-1} + delta*B[t,n]*u;  y = sum_n h*C[t,n].
// Epilogue fused: y = (y + u*D[d]) * silu(res); writes y as bf16 for out_proj.
// ---------------------------------------------------------------------------
__global__ __launch_bounds__(256) void ssm_scan_kernel(
    __bf16* __restrict__ yb, const float* __restrict__ delta,
    const float* __restrict__ xc, const float* __restrict__ ssm,
    const float* __restrict__ xr, const float* __restrict__ A_log,
    const float* __restrict__ Dp)
{
    int tid = blockIdx.x * blockDim.x + threadIdx.x;   // 0 .. B*D*16-1
    int n   = tid & 15;
    int bd  = tid >> 4;                                // 0..2047
    int b   = bd >> 10;
    int d   = bd & (D_SZ - 1);

    const float A  = -__expf(A_log[n]);
    const float Dd = Dp[d];
    float h = 0.0f;
    const size_t rbase = (size_t)b * T_SZ;

    for (int t = 0; t < T_SZ; ++t) {
        size_t bt = rbase + t;
        float dl = delta[bt * D_SZ + d];
        float u  = xc[bt * D_SZ + d];
        float Bv = ssm[bt * SSM_LD + DTRANK + n];           // B params
        float Cv = ssm[bt * SSM_LD + DTRANK + NSTATE + n];  // C params
        float dA = __expf(dl * A);
        h = dA * h + dl * Bv * u;
        float p = h * Cv;
        // reduce the 16 states sharing (b,d): lane-permute tree
        p += __shfl_xor(p, 1, 32);
        p += __shfl_xor(p, 2, 32);
        p += __shfl_xor(p, 4, 32);
        p += __shfl_xor(p, 8, 32);
        if (n == 0) {
            float res = xr[bt * 2048 + D_SZ + d];           // gate branch
            float g   = res / (1.0f + __expf(-res));
            float val = (p + u * Dd) * g;
            yb[bt * D_SZ + d] = (__bf16)val;
        }
    }
}

// ---------------------------------------------------------------------------
// Host-side orchestration
// ---------------------------------------------------------------------------
extern "C" void kernel_launch(void* const* d_in, const int* in_sizes, int n_in,
                              void* d_out, int out_size, void* d_ws, size_t ws_size,
                              hipStream_t stream) {
    const float* x         = (const float*)d_in[0];
    const float* in_w      = (const float*)d_in[1];
    const float* in_b      = (const float*)d_in[2];
    const float* conv_w    = (const float*)d_in[3];
    const float* conv_b    = (const float*)d_in[4];
    const float* xp_w      = (const float*)d_in[5];
    const float* xp_b      = (const float*)d_in[6];
    const float* dt_w      = (const float*)d_in[7];
    const float* dt_b      = (const float*)d_in[8];
    const float* A_log     = (const float*)d_in[9];
    const float* D_param   = (const float*)d_in[10];
    const float* out_w     = (const float*)d_in[11];
    const float* out_b     = (const float*)d_in[12];
    const float* skip_w    = (const float*)d_in[13];
    const float* skip_b    = (const float*)d_in[14];
    float* out = (float*)d_out;

    // workspace carve-out
    char* wp = (char*)d_ws;
    auto carve = [&](size_t bytes) -> void* {
        void* p = (void*)wp;
        wp += (bytes + 255) & ~(size_t)255;
        return p;
    };
    __bf16* xb      = (__bf16*)carve((size_t)BT_SZ * D_SZ * 2);       // x bf16
    __bf16* w_in_b  = (__bf16*)carve((size_t)2048 * 1024 * 2);
    __bf16* w_xp_b  = (__bf16*)carve((size_t)SSM_LD * 1024 * 2);      // padded 96->128 rows
    __bf16* w_dt_b  = (__bf16*)carve((size_t)1024 * 64 * 2);
    __bf16* w_out_b = (__bf16*)carve((size_t)1024 * 1024 * 2);
    __bf16* w_sk_b  = (__bf16*)carve((size_t)1024 * 1024 * 2);
    float*  xr      = (float*) carve((size_t)BT_SZ * 2048 * 4);       // in_proj output
    float*  xc      = (float*) carve((size_t)BT_SZ * D_SZ * 4);       // conv+silu
    __bf16* xcb     = (__bf16*)carve((size_t)BT_SZ * D_SZ * 2);
    float*  ssm     = (float*) carve((size_t)BT_SZ * SSM_LD * 4);
    __bf16* dlowb   = (__bf16*)carve((size_t)BT_SZ * 64 * 2);
    float*  delta   = (float*) carve((size_t)BT_SZ * D_SZ * 4);       // softplus(dt)
    __bf16* yb      = (__bf16*)carve((size_t)BT_SZ * D_SZ * 2);
    (void)ws_size; (void)in_sizes; (void)n_in; (void)out_size;

    const int TPB = 256;
    auto blks = [](int n, int tpb) { return (n + tpb - 1) / tpb; };

    // 1) bf16 casts of activations and weights
    cvt_f32_bf16<<<blks(BT_SZ * D_SZ, TPB), TPB, 0, stream>>>(xb, x, BT_SZ * D_SZ);
    cvt_f32_bf16<<<blks(2048 * 1024, TPB), TPB, 0, stream>>>(w_in_b, in_w, 2048 * 1024);
    cvt_pad_rows<<<blks(SSM_LD * 1024, TPB), TPB, 0, stream>>>(w_xp_b, xp_w, SSM_LD, 96, 1024);
    cvt_f32_bf16<<<blks(1024 * 64, TPB), TPB, 0, stream>>>(w_dt_b, dt_w, 1024 * 64);
    cvt_f32_bf16<<<blks(1024 * 1024, TPB), TPB, 0, stream>>>(w_out_b, out_w, 1024 * 1024);
    cvt_f32_bf16<<<blks(1024 * 1024, TPB), TPB, 0, stream>>>(w_sk_b, skip_w, 1024 * 1024);

    // 2) in_proj: xr = x @ in_w^T + in_b     (M=4096, N=2048, K=1024)
    gemm_bf16_wmma<<<dim3(2048 / 64, BT_SZ / 128), 128, 0, stream>>>(
        xr, xb, w_in_b, in_b, BT_SZ, 2048, 1024, 1024, 1024, 2048, 2048, 0);

    // 3) depthwise causal conv + silu -> xc / xcb
    conv_silu_kernel<<<blks(BT_SZ * D_SZ, TPB), TPB, 0, stream>>>(xc, xcb, xr, conv_w, conv_b);

    // 4) x_proj: ssm = xc @ xp_w^T + xp_b    (N padded to 128, bias valid for col<96)
    gemm_bf16_wmma<<<dim3(SSM_LD / 64, BT_SZ / 128), 128, 0, stream>>>(
        ssm, xcb, w_xp_b, xp_b, BT_SZ, SSM_LD, 1024, 1024, 1024, SSM_LD, 96, 0);

    // 5) bf16 cast of delta-rank slice ssm[:, 0:64]
    cvt_strided<<<blks(BT_SZ * 64, TPB), TPB, 0, stream>>>(dlowb, ssm, BT_SZ, 64, SSM_LD);

    // 6) dt_proj + softplus: delta = softplus(dlow @ dt_w^T + dt_b)  (K=64)
    gemm_bf16_wmma<<<dim3(1024 / 64, BT_SZ / 128), 128, 0, stream>>>(
        delta, dlowb, w_dt_b, dt_b, BT_SZ, 1024, 64, 64, 64, 1024, 1024, 1);

    // 7) selective scan + D skip + gate:  yb = ((scan) + u*D) * silu(res)
    ssm_scan_kernel<<<blks(B_SZ * D_SZ * NSTATE, TPB), TPB, 0, stream>>>(
        yb, delta, xc, ssm, xr, A_log, D_param);

    // 8) out = y @ out_w^T + out_b + x @ skip_w^T + skip_b
    gemm_bf16_wmma<<<dim3(1024 / 64, BT_SZ / 128), 128, 0, stream>>>(
        out, xb, w_sk_b, skip_b, BT_SZ, 1024, 1024, 1024, 1024, 1024, 1024, 0);
    gemm_bf16_wmma<<<dim3(1024 / 64, BT_SZ / 128), 128, 0, stream>>>(
        out, yb, w_out_b, out_b, BT_SZ, 1024, 1024, 1024, 1024, 1024, 1024, 2);
}
